// SpatialGCNAttentionLayer_52115133170283
// MI455X (gfx1250) — matrix operand order repaired and checked by reading, not compiled, not run on hardware
//
#include <hip/hip_runtime.h>

// SpatialGCNAttentionLayer for MI455X (gfx1250), wave32 + WMMA bf16 + TDM.
//
// B=8, N=256, D=128, E=32, H=4, HD=32.
// Algebra: edge-conditioned K/V folded into per-(i,h) E=32 vectors:
//   score = q.Kh_j + (Wke^T q).ef_ij     (k bias dropped: softmax shift-inv)
//   o     = attn.Vh + Wve.(attn.ef) + (wv.edge_b + bv)   (sum attn = 1)
// No (B,N,N,D) tensor is ever formed; edge_features (67 MB) is streamed once,
// HBM roofline ~3us. ef tiles are DMA'd into LDS by the Tensor Data Mover
// (with TDM row padding -> conflict-free 33-dword LDS stride), overlapped with
// the Q.Kh^T WMMA pass.

constexpr int kB = 8, kN = 256, kD = 128, kE = 32;
constexpr int EFW = 33;            // LDS row stride (f32) for ef tiles (TDM pad)
constexpr int SCW = 260;           // padded scores row stride (f32)
constexpr int OBW = 132;           // padded o_base row stride (f32)
constexpr float kQScale = 0.17677669529663687f;   // 1/sqrt(32)

typedef __attribute__((ext_vector_type(16))) __bf16 v16bf;
typedef __attribute__((ext_vector_type(8)))  float  v8f;
typedef __attribute__((ext_vector_type(4)))  unsigned int v4u;
typedef __attribute__((ext_vector_type(8)))  int v8i;
typedef __attribute__((ext_vector_type(4)))  int v4i;

#if __has_builtin(__builtin_amdgcn_tensor_load_to_lds) && \
    __has_builtin(__builtin_amdgcn_s_wait_tensorcnt)
#define HAVE_TDM 1
#else
#define HAVE_TDM 0
#endif

union BF16x16 { v16bf v; __bf16 e[16]; };

__device__ __forceinline__ v8f wmma_bf16(v16bf a, v16bf b, v8f c) {
  return __builtin_amdgcn_wmma_f32_16x16x32_bf16(false, a, false, b, (short)0, c,
                                                 false, false);
}

// ---- fragment builders ------------------------------------------------------
// A (16xK, row m = lane%16, group g = lane/16): pair v holds K = 2v+8g (+8 if v>=4)
// B (Kx16, col n = lane%16):                    pair v holds K = 2v+8g (+8 if v>=4)

__device__ __forceinline__ v16bf loadA_f32(const float* src, int stride,
                                           int k0, int lane) {
  int g = lane >> 4, m = lane & 15;
  BF16x16 r;
#pragma unroll
  for (int v = 0; v < 8; ++v) {
    int k = k0 + 2 * v + 8 * g + (v >= 4 ? 8 : 0);
    r.e[2 * v]     = (__bf16)src[m * stride + k];
    r.e[2 * v + 1] = (__bf16)src[m * stride + k + 1];
  }
  return r.v;
}

__device__ __forceinline__ v16bf loadA_bf16(const __bf16* src, int stride,
                                            int k0, int lane) {
  int g = lane >> 4, m = lane & 15;
  BF16x16 r;
#pragma unroll
  for (int v = 0; v < 8; ++v) {
    int k = k0 + 2 * v + 8 * g + (v >= 4 ? 8 : 0);
    r.e[2 * v]     = src[(size_t)m * stride + k];
    r.e[2 * v + 1] = src[(size_t)m * stride + k + 1];
  }
  return r.v;
}

// B[k][n] = src[(n0+n)*stride + k0+k]  (contiguous along k)
__device__ __forceinline__ v16bf loadB_rowN_bf16(const __bf16* src, int stride,
                                                 int k0, int n0, int lane) {
  int g = lane >> 4, n = (lane & 15) + n0;
  BF16x16 r;
#pragma unroll
  for (int v = 0; v < 8; ++v) {
    int k = k0 + 2 * v + 8 * g + (v >= 4 ? 8 : 0);
    r.e[2 * v]     = src[(size_t)n * stride + k];
    r.e[2 * v + 1] = src[(size_t)n * stride + k + 1];
  }
  return r.v;
}

__device__ __forceinline__ v16bf loadB_rowN_f32(const float* src, int stride,
                                                int k0, int n0, int lane) {
  int g = lane >> 4, n = (lane & 15) + n0;
  BF16x16 r;
#pragma unroll
  for (int v = 0; v < 8; ++v) {
    int k = k0 + 2 * v + 8 * g + (v >= 4 ? 8 : 0);
    r.e[2 * v]     = (__bf16)src[(size_t)n * stride + k];
    r.e[2 * v + 1] = (__bf16)src[(size_t)n * stride + k + 1];
  }
  return r.v;
}

// B[k][n] = src[(k0+k)*stride + n0+n]  (gather along k)
__device__ __forceinline__ v16bf loadB_rowK_f32(const float* src, int stride,
                                                int k0, int n0, int lane) {
  int g = lane >> 4, n = (lane & 15) + n0;
  BF16x16 r;
#pragma unroll
  for (int v = 0; v < 8; ++v) {
    int k = k0 + 2 * v + 8 * g + (v >= 4 ? 8 : 0);
    r.e[2 * v]     = (__bf16)src[(size_t)k * stride + n];
    r.e[2 * v + 1] = (__bf16)src[(size_t)(k + 1) * stride + n];
  }
  return r.v;
}

// ---- kernel 0: fold weights -------------------------------------------------
__global__ __launch_bounds__(256) void k0_prep(
    const float* __restrict__ wq, const float* __restrict__ wk,
    const float* __restrict__ wv, const float* __restrict__ gcn_w,
    const float* __restrict__ out_w, const float* __restrict__ edge_w,
    const float* __restrict__ edge_b, const float* __restrict__ bv,
    __bf16* wq_bf, __bf16* wk_bf, __bf16* wv_bf, __bf16* gcnw_bf,
    __bf16* outw_bf, float* wke, float* wve, float* vb2) {
  int t = threadIdx.x;
  for (int i = t; i < kD * kD; i += 256) {
    wq_bf[i]   = (__bf16)wq[i];
    wk_bf[i]   = (__bf16)wk[i];
    wv_bf[i]   = (__bf16)wv[i];
    gcnw_bf[i] = (__bf16)gcn_w[i];
    outw_bf[i] = (__bf16)out_w[i];
  }
  for (int i = t; i < kD * kE; i += 256) {
    int d = i >> 5, e = i & 31;
    float sk = 0.f, sv = 0.f;
    for (int d2 = 0; d2 < kD; ++d2) {
      float ew = edge_w[d2 * kE + e];
      sk += wk[d * kD + d2] * ew;
      sv += wv[d * kD + d2] * ew;
    }
    wke[i] = sk;
    wve[i] = sv;
  }
  for (int d = t; d < kD; d += 256) {
    float s = 0.f;
    for (int d2 = 0; d2 < kD; ++d2) s += wv[d * kD + d2] * edge_b[d2];
    vb2[d] = s + bv[d];
  }
}

// ---- kernel 1: Q / Kh / Vh projections + transposed h copy ------------------
__global__ __launch_bounds__(256) void k1_proj(
    const float* __restrict__ h, const float* __restrict__ bq,
    const __bf16* __restrict__ wq_bf, const __bf16* __restrict__ wk_bf,
    const __bf16* __restrict__ wv_bf, float* q_out, __bf16* kh_out,
    __bf16* vhT_out, __bf16* hT_out) {
  __shared__ float hs[16][132];
  int t = threadIdx.x;
  int r0 = blockIdx.x * 16;       // global row = b*N + node
  int b = r0 >> 8, n0 = r0 & 255;
  for (int idx = t; idx < 16 * 128; idx += 256) {
    int m = idx >> 7, d = idx & 127;
    float v = h[(size_t)(r0 + m) * kD + d];
    hs[m][d] = v;
    hT_out[((size_t)b * kD + d) * kN + n0 + m] = (__bf16)v;
  }
  __syncthreads();
  int lane = t & 31, wave = t >> 5;
  v16bf a[4];
#pragma unroll
  for (int kc = 0; kc < 4; ++kc) a[kc] = loadA_f32(&hs[0][0], 132, kc * 32, lane);
  int d0 = wave * 16;
  v8f cq = {}, ck = {}, cv = {};
#pragma unroll
  for (int kc = 0; kc < 4; ++kc) {
    cq = wmma_bf16(a[kc], loadB_rowN_bf16(wq_bf, kD, kc * 32, d0, lane), cq);
    ck = wmma_bf16(a[kc], loadB_rowN_bf16(wk_bf, kD, kc * 32, d0, lane), ck);
    cv = wmma_bf16(a[kc], loadB_rowN_bf16(wv_bf, kD, kc * 32, d0, lane), cv);
  }
  int g = lane >> 4, nn = lane & 15;
#pragma unroll
  for (int r = 0; r < 8; ++r) {
    int m = r + 8 * g, d = d0 + nn;
    q_out[(size_t)(r0 + m) * kD + d]  = cq[r] + bq[d];
    kh_out[(size_t)(r0 + m) * kD + d] = (__bf16)ck[r];
    vhT_out[((size_t)b * kD + d) * kN + n0 + m] = (__bf16)cv[r];
  }
}

// ---- kernel 2: fused masked multi-head attention (4 query nodes / block) ----
__global__ __launch_bounds__(256) void k2_attn(
    const float* __restrict__ adj, const float* __restrict__ srcm,
    const float* __restrict__ ef, const float* __restrict__ q_ws,
    const __bf16* __restrict__ kh, const __bf16* __restrict__ vhT,
    const float* __restrict__ wke, const float* __restrict__ wve,
    const float* __restrict__ vb2, __bf16* o_out) {
  extern __shared__ char smem[];
  float* eff    = (float*)smem;              // [4][256][EFW]  (TDM-padded f32)
  float* scores = eff + 4 * 256 * EFW;       // [16][SCW]
  float* obase  = scores + 16 * SCW;         // [16][OBW]
  float* efa    = obase + 16 * OBW;          // [16][32]
  float* qs     = efa + 16 * 32;             // [4][128] (pre-scaled)
  float* qe     = qs + 4 * 128;              // [16][32]

  int t = threadIdx.x, lane = t & 31, wave = t >> 5;
  int b = blockIdx.x >> 6;
  int i0 = (blockIdx.x & 63) * 4;

#if HAVE_TDM
  // Tensor Data Mover: DMA four 256x32 f32 tiles HBM -> LDS, padding each
  // 32-dword row by 1 dword (pad_interval=32dw, pad_amount=1dw) -> stride EFW.
  if (wave == 0) {
    for (int il = 0; il < 4; ++il) {
      unsigned long long ga =
          (unsigned long long)(const void*)(ef +
              (((size_t)b * kN + i0 + il) * kN) * kE);
      unsigned lds = (unsigned)(size_t)(void*)(eff + il * 256 * EFW);
      v4u g0 = { 1u,                                   // count=1 user desc
                 lds,                                  // lds_addr
                 (unsigned)(ga & 0xffffffffu),         // global_addr[31:0]
                 (unsigned)((ga >> 32) & 0x01ffffffu)  // global_addr[56:32]
                     | 0x80000000u };                  // type=2 (image)
      v8i g1 = { (int)0x01120000,   // data_size=4B, pad_en, intv=32dw, amt=1dw
                 (int)(32u << 16),  // tensor_dim0 = 32
                 (int)(256u << 16), // tensor_dim1 = 256
                 (int)(32u << 16),  // tile_dim0 = 32
                 (int)256,          // tile_dim1 = 256, tile_dim2 = 0
                 (int)32,           // tensor_dim0_stride = 32
                 0, 0 };
      v4i gz = { 0, 0, 0, 0 };
#if __has_include(<hip/amd_detail/amd_gfx1250_TDM.h>)
      v8i gz8 = { 0, 0, 0, 0, 0, 0, 0, 0 };
      __builtin_amdgcn_tensor_load_to_lds(g0, g1, gz, gz, gz8, 0);
#else
      __builtin_amdgcn_tensor_load_to_lds(g0, g1, gz, gz, 0);
#endif
    }
  }
#else
  for (int idx = t; idx < 4 * 256 * 8; idx += 256) {
    int c4 = idx & 7, j = (idx >> 3) & 255, il = idx >> 11;
    const float4 v = ((const float4*)(ef +
        (((size_t)b * kN + i0 + il) * kN + j) * kE))[c4];
    *(float4*)&eff[(il * 256 + j) * EFW + c4 * 4] = v;
  }
#endif

  for (int idx = t; idx < 4 * 128; idx += 256)
    qs[idx] = q_ws[((size_t)b * kN + i0 + (idx >> 7)) * kD + (idx & 127)] * kQScale;
  __syncthreads();

  // qe[(il,h)][e] = sum_dh qs[il][h*32+dh] * wke[h*32+dh][e]
  for (int idx = t; idx < 16 * 32; idx += 256) {
    int e = idx & 31, row = idx >> 5, il = row >> 2, hh = row & 3;
    float s = 0.f;
#pragma unroll 8
    for (int dh = 0; dh < 32; ++dh)
      s += qs[il * 128 + hh * 32 + dh] * wke[(hh * 32 + dh) * kE + e];
    qe[row * 32 + e] = s;
  }
  __syncthreads();

  // ---- s0 = (masked-per-head Q) . Kh^T : 16 rows x 256 j  (ef-independent,
  //      overlaps the TDM transfer)
  {
    v16bf aq[4];
#pragma unroll
    for (int kc = 0; kc < 4; ++kc) {
      int g = lane >> 4, m = lane & 15, il = m >> 2, hh = m & 3;
      BF16x16 r;
#pragma unroll
      for (int v = 0; v < 8; ++v) {
        int k = 2 * v + 8 * g + (v >= 4 ? 8 : 0);
        float v0 = (hh == kc) ? qs[il * 128 + kc * 32 + k] : 0.f;
        float v1 = (hh == kc) ? qs[il * 128 + kc * 32 + k + 1] : 0.f;
        r.e[2 * v] = (__bf16)v0;
        r.e[2 * v + 1] = (__bf16)v1;
      }
      aq[kc] = r.v;
    }
    const __bf16* khb = kh + (size_t)b * kN * kD;
    for (int jt = wave; jt < 16; jt += 8) {
      int j0 = jt * 16;
      v8f c = {};
#pragma unroll
      for (int kc = 0; kc < 4; ++kc)
        c = wmma_bf16(aq[kc], loadB_rowN_bf16(khb, kD, kc * 32, j0, lane), c);
      int g = lane >> 4, nn = lane & 15;
#pragma unroll
      for (int r = 0; r < 8; ++r) scores[(r + 8 * g) * SCW + j0 + nn] = c[r];
    }
  }
  __syncthreads();
#if HAVE_TDM
  if (wave == 0) __builtin_amdgcn_s_wait_tensorcnt(0);
  __syncthreads();
#endif

  // ---- edge scores: per i, qe (4x32) . ef^T  (M=16 tile, 4 rows used)
  for (int tile = wave; tile < 64; tile += 8) {
    int il = tile >> 4, j0 = (tile & 15) * 16;
    int g = lane >> 4, m = lane & 15;
    BF16x16 ra;
#pragma unroll
    for (int v = 0; v < 8; ++v) {
      int k = 2 * v + 8 * g + (v >= 4 ? 8 : 0);
      float v0 = (m < 4) ? qe[(il * 4 + m) * 32 + k] : 0.f;
      float v1 = (m < 4) ? qe[(il * 4 + m) * 32 + k + 1] : 0.f;
      ra.e[2 * v] = (__bf16)v0;
      ra.e[2 * v + 1] = (__bf16)v1;
    }
    v16bf bb = loadB_rowN_f32(eff + il * 256 * EFW, EFW, 0, j0, lane);
    v8f c = {};
    c = wmma_bf16(ra.v, bb, c);
    int nn = lane & 15;
    if ((lane >> 4) == 0) {
#pragma unroll
      for (int r = 0; r < 4; ++r) scores[(il * 4 + r) * SCW + j0 + nn] += c[r];
    }
  }
  __syncthreads();

  // mask where adj*src_mask == 0
  for (int idx = t; idx < 16 * 256; idx += 256) {
    int j = idx & 255, row = idx >> 8, il = row >> 2;
    size_t base = ((size_t)b * kN + i0 + il) * kN + j;
    if (adj[base] * srcm[base] == 0.f) scores[row * SCW + j] = -1e30f;
  }
  __syncthreads();

  // softmax per row (wave32 shuffle reduction)
  for (int row = wave; row < 16; row += 8) {
    float mx = -1e30f;
    for (int j = lane; j < 256; j += 32) mx = fmaxf(mx, scores[row * SCW + j]);
#pragma unroll
    for (int s = 16; s; s >>= 1) mx = fmaxf(mx, __shfl_xor(mx, s, 32));
    float sum = 0.f;
    for (int j = lane; j < 256; j += 32) {
      float p = __expf(scores[row * SCW + j] - mx);
      scores[row * SCW + j] = p;
      sum += p;
    }
#pragma unroll
    for (int s = 16; s; s >>= 1) sum += __shfl_xor(sum, s, 32);
    float inv = 1.f / sum;
    for (int j = lane; j < 256; j += 32) scores[row * SCW + j] *= inv;
  }
  __syncthreads();

  // ---- o_base = attn . Vh  (wave -> one d-ntile)
  {
    int d0 = wave * 16;
    const __bf16* vhb = vhT + (size_t)b * kD * kN;
    v8f c = {};
#pragma unroll
    for (int kc = 0; kc < 8; ++kc) {
      v16bf aa = loadA_f32(scores, SCW, kc * 32, lane);
      v16bf bb = loadB_rowN_bf16(vhb, kN, kc * 32, d0, lane);
      c = wmma_bf16(aa, bb, c);
    }
    int g = lane >> 4, nn = lane & 15;
#pragma unroll
    for (int r = 0; r < 8; ++r) obase[(r + 8 * g) * OBW + d0 + nn] = c[r];
  }
  // ---- efa = attn . ef  (per i: 4 rows x 32 e; wave -> (il, e-tile))
  {
    int il = wave >> 1, e0 = (wave & 1) * 16;
    v8f c = {};
#pragma unroll
    for (int kc = 0; kc < 8; ++kc) {
      int g = lane >> 4, m = lane & 15;
      BF16x16 ra;
#pragma unroll
      for (int v = 0; v < 8; ++v) {
        int k = kc * 32 + 2 * v + 8 * g + (v >= 4 ? 8 : 0);
        float v0 = (m < 4) ? scores[(il * 4 + m) * SCW + k] : 0.f;
        float v1 = (m < 4) ? scores[(il * 4 + m) * SCW + k + 1] : 0.f;
        ra.e[2 * v] = (__bf16)v0;
        ra.e[2 * v + 1] = (__bf16)v1;
      }
      v16bf bb = loadB_rowK_f32(eff + il * 256 * EFW, EFW, kc * 32, e0, lane);
      c = wmma_bf16(ra.v, bb, c);
    }
    int nn = lane & 15;
    if ((lane >> 4) == 0) {
#pragma unroll
      for (int r = 0; r < 4; ++r) efa[(il * 4 + r) * 32 + e0 + nn] = c[r];
    }
  }
  __syncthreads();

  // o = o_base + Wve . efa + vb2   (head picked by d>>5)
  for (int idx = t; idx < 4 * 128; idx += 256) {
    int d = idx & 127, il = idx >> 7;
    int row = il * 4 + (d >> 5);
    float s = obase[row * OBW + d] + vb2[d];
#pragma unroll 8
    for (int e = 0; e < 32; ++e) s += wve[d * kE + e] * efa[row * 32 + e];
    o_out[((size_t)b * kN + i0 + il) * kD + d] = (__bf16)s;
  }
}

// ---- kernel 3: GCN branch + out-proj + residual + LayerNorm + combine -------
__global__ __launch_bounds__(256) void k3_final(
    const float* __restrict__ h, const float* __restrict__ adj,
    const __bf16* __restrict__ hT, const __bf16* __restrict__ o_bf,
    const __bf16* __restrict__ gcnw_bf, const __bf16* __restrict__ outw_bf,
    const float* __restrict__ gcn_b, const float* __restrict__ out_b,
    const float* __restrict__ ln_g, const float* __restrict__ ln_b,
    float* __restrict__ out) {
  __shared__ float xs[16][128];
  __shared__ float aggs[16][132];
  __shared__ float hgcn[16][128];
  __shared__ float deg[16];
  __shared__ float mu[16], rstd[16];
  int t = threadIdx.x, lane = t & 31, wave = t >> 5;
  int b = blockIdx.x >> 4;
  int i0 = (blockIdx.x & 15) * 16;
  size_t r0 = (size_t)b * kN + i0;

  if (t < 16) deg[t] = 1.f;    // self-loop from the added eye
  __syncthreads();
  {  // parallel degree: 16 threads per row, ds_add_f32 reduction
    int row = t >> 4;
    const float* arow = adj + ((size_t)b * kN + i0 + row) * kN;
    float s = 0.f;
    for (int j = (t & 15); j < kN; j += 16) s += arow[j];
    atomicAdd(&deg[row], s);
  }
  // GCN aggregation: (adj + I) @ h   via WMMA, wave -> d-ntile
  {
    int d0 = wave * 16;
    const __bf16* hTb = hT + (size_t)b * kD * kN;
    int g = lane >> 4, m = lane & 15;
    const float* arow = adj + ((size_t)b * kN + i0 + m) * kN;
    v8f c = {};
#pragma unroll
    for (int kc = 0; kc < 8; ++kc) {
      BF16x16 ra;
#pragma unroll
      for (int v = 0; v < 8; ++v) {
        int k = kc * 32 + 2 * v + 8 * g + (v >= 4 ? 8 : 0);
        float v0 = arow[k] + ((k == i0 + m) ? 1.f : 0.f);
        float v1 = arow[k + 1] + ((k + 1 == i0 + m) ? 1.f : 0.f);
        ra.e[2 * v] = (__bf16)v0;
        ra.e[2 * v + 1] = (__bf16)v1;
      }
      c = wmma_bf16(ra.v, loadB_rowN_bf16(hTb, kN, kc * 32, d0, lane), c);
    }
    int nn = lane & 15;
#pragma unroll
    for (int r = 0; r < 8; ++r) aggs[r + 8 * g][d0 + nn] = c[r];
  }
  __syncthreads();
  for (int idx = t; idx < 16 * 128; idx += 256) {
    int m = idx >> 7, d = idx & 127;
    aggs[m][d] /= deg[m];
  }
  __syncthreads();
  // GCN projection and attention output projection
  {
    int d0 = wave * 16;
    v8f cg = {}, co = {};
#pragma unroll
    for (int kc = 0; kc < 4; ++kc) {
      v16bf ag = loadA_f32(&aggs[0][0], 132, kc * 32, lane);
      cg = wmma_bf16(ag, loadB_rowN_bf16(gcnw_bf, kD, kc * 32, d0, lane), cg);
      v16bf ao = loadA_bf16(o_bf + r0 * kD, kD, kc * 32, lane);
      co = wmma_bf16(ao, loadB_rowN_bf16(outw_bf, kD, kc * 32, d0, lane), co);
    }
    int g = lane >> 4, nn = lane & 15;
#pragma unroll
    for (int r = 0; r < 8; ++r) {
      int m = r + 8 * g, d = d0 + nn;
      xs[m][d] = h[(r0 + m) * kD + d] + co[r] + out_b[d];
      hgcn[m][d] = cg[r] + gcn_b[d];
    }
  }
  __syncthreads();
  // LayerNorm stats per row
  for (int row = wave; row < 16; row += 8) {
    float s = 0.f;
    for (int d = lane; d < 128; d += 32) s += xs[row][d];
#pragma unroll
    for (int sh = 16; sh; sh >>= 1) s += __shfl_xor(s, sh, 32);
    float m_ = s * (1.f / 128.f);
    float v = 0.f;
    for (int d = lane; d < 128; d += 32) {
      float z = xs[row][d] - m_;
      v += z * z;
    }
#pragma unroll
    for (int sh = 16; sh; sh >>= 1) v += __shfl_xor(v, sh, 32);
    if (lane == 0) {
      mu[row] = m_;
      rstd[row] = rsqrtf(v * (1.f / 128.f) + 1e-5f);
    }
  }
  __syncthreads();
  for (int idx = t; idx < 16 * 128; idx += 256) {
    int m = idx >> 7, d = idx & 127;
    float ha = ln_g[d] * (xs[m][d] - mu[m]) * rstd[m] + ln_b[d];
    out[(r0 + m) * kD + d] = hgcn[m][d] + ha;
  }
}

// ---- host side --------------------------------------------------------------
extern "C" void kernel_launch(void* const* d_in, const int* in_sizes, int n_in,
                              void* d_out, int out_size, void* d_ws, size_t ws_size,
                              hipStream_t stream) {
  const float* h      = (const float*)d_in[0];
  const float* adj    = (const float*)d_in[1];
  const float* ef     = (const float*)d_in[2];
  const float* srcm   = (const float*)d_in[3];
  const float* gcn_w  = (const float*)d_in[4];
  const float* gcn_b  = (const float*)d_in[5];
  const float* edge_w = (const float*)d_in[6];
  const float* edge_b = (const float*)d_in[7];
  const float* wq     = (const float*)d_in[8];
  const float* wk     = (const float*)d_in[9];
  const float* wv     = (const float*)d_in[10];
  const float* bq     = (const float*)d_in[11];
  // d_in[12] = bk: dropped (softmax shift invariance)
  const float* bv     = (const float*)d_in[13];
  const float* out_w  = (const float*)d_in[14];
  const float* out_b  = (const float*)d_in[15];
  const float* ln_g   = (const float*)d_in[16];
  const float* ln_b   = (const float*)d_in[17];
  float* out = (float*)d_out;

  char* ws = (char*)d_ws;
  float*  q_ws    = (float*)(ws);                    // 1,048,576 B
  __bf16* kh_ws   = (__bf16*)(ws + 1048576);         //   524,288 B
  __bf16* vhT_ws  = (__bf16*)(ws + 1572864);         //   524,288 B
  __bf16* hT_ws   = (__bf16*)(ws + 2097152);         //   524,288 B
  __bf16* o_ws    = (__bf16*)(ws + 2621440);         //   524,288 B
  __bf16* wq_bf   = (__bf16*)(ws + 3145728);
  __bf16* wk_bf   = (__bf16*)(ws + 3178496);
  __bf16* wv_bf   = (__bf16*)(ws + 3211264);
  __bf16* gcnw_bf = (__bf16*)(ws + 3244032);
  __bf16* outw_bf = (__bf16*)(ws + 3276800);
  float*  wke     = (float*)(ws + 3309568);
  float*  wve     = (float*)(ws + 3325952);
  float*  vb2     = (float*)(ws + 3342336);

  k0_prep<<<1, 256, 0, stream>>>(wq, wk, wv, gcn_w, out_w, edge_w, edge_b, bv,
                                 wq_bf, wk_bf, wv_bf, gcnw_bf, outw_bf,
                                 wke, wve, vb2);
  k1_proj<<<(kB * kN) / 16, 256, 0, stream>>>(h, bq, wq_bf, wk_bf, wv_bf,
                                              q_ws, kh_ws, vhT_ws, hT_ws);
  constexpr size_t K2_LDS = (size_t)4 * 256 * EFW * 4   // ef tiles (f32, padded)
                          + 16 * SCW * 4                // scores/attn
                          + 16 * OBW * 4                // o_base
                          + 16 * 32 * 4                 // efa
                          + 4 * 128 * 4                 // q (scaled)
                          + 16 * 32 * 4;                // qe
  k2_attn<<<kB * (kN / 4), 256, K2_LDS, stream>>>(adj, srcm, ef, q_ws, kh_ws,
                                                  vhT_ws, wke, wve, vb2, o_ws);
  k3_final<<<kB * (kN / 16), 256, 0, stream>>>(h, adj, hT_ws, o_ws, gcnw_bf,
                                               outw_bf, gcn_b, out_b, ln_g,
                                               ln_b, out);
  (void)in_sizes; (void)n_in; (void)out_size; (void)ws_size;
}
